// Transformer_19971597926919
// MI455X (gfx1250) — matrix operand order repaired and checked by reading, not compiled, not run on hardware
//
#include <hip/hip_runtime.h>

// ---------------------------------------------------------------------------
// Transformer encoder-decoder forward for MI455X (gfx1250, wave32, WMMA).
// All GEMMs run on v_wmma_f32_16x16x32_f16 with f32 accumulation; bias,
// residual, ReLU and score-scale are fused into the GEMM epilogue.
// Double-buffered LDS pipeline: next K-tile's global loads issued before the
// current tile's WMMAs so HBM latency hides under matrix ops.
// ---------------------------------------------------------------------------

typedef _Float16     v16h __attribute__((ext_vector_type(16)));
typedef float        v8f  __attribute__((ext_vector_type(8)));
typedef float        v4f  __attribute__((ext_vector_type(4)));
typedef unsigned int v4u  __attribute__((ext_vector_type(4)));

union Frag16 { v16h h; v4u q[2]; };

#define B_SZ   4
#define L_SZ   512
#define D_SZ   512
#define H_SZ   8
#define HD_SZ  64
#define FF_SZ  2048
#define V_SZ   32000
#define M_SZ   (B_SZ * L_SZ)   // 2048 rows of activations
#define LPAD   40              // LDS row stride in halves (80B, 16B-aligned)

// ---------------------------------------------------------------------------
// Generic batched GEMM:  C = alpha * (A @ B) + bias + residual, optional ReLU
//   A: [M,K] row-major (lda).  B: if transB, [N,K] (ldb); else [K,N] (ldb).
//   Batch z -> (z1 = z/bdiv, z2 = z%bdiv) with per-operand strides.
//   Grid: (N/64, M/64, nbatch), 128 threads = 4 waves, 32x32 per wave.
// ---------------------------------------------------------------------------
__global__ __launch_bounds__(128) void gemm_wmma_f16(
    const float* __restrict__ A, const float* __restrict__ Bm,
    const float* __restrict__ bias, const float* __restrict__ Rs,
    float* __restrict__ C,
    int K, int lda, int ldb, int ldc,
    long sA1, long sA2, long sB1, long sB2, long sC1, long sC2,
    int bdiv, int transB, float alpha, int relu)
{
  const int z  = blockIdx.z;
  const int z1 = z / bdiv, z2 = z - z1 * bdiv;
  A  += (long)z1 * sA1 + (long)z2 * sA2;
  Bm += (long)z1 * sB1 + (long)z2 * sB2;
  C  += (long)z1 * sC1 + (long)z2 * sC2;

  const int tile_m = blockIdx.y << 6;
  const int tile_n = blockIdx.x << 6;

  // double-buffered LDS tiles (2 * 2 * 64 * 40 * 2B = 20 KB of 320 KB/WGP)
  __shared__ __attribute__((aligned(16))) _Float16 As[2][64][LPAD];
  __shared__ __attribute__((aligned(16))) _Float16 Bs[2][64][LPAD]; // Bs[.][n][k]

  const int tid  = threadIdx.x;
  const int lane = tid & 31;
  const int wid  = tid >> 5;
  const int wrow = (wid >> 1) << 5;   // 0 / 32
  const int wcol = (wid & 1) << 5;    // 0 / 32
  const int l16  = lane & 15;
  const int lhi  = lane >> 4;

  v8f acc[2][2] = {};
  v4f ra[4], rb[4];   // staged registers: 8 K-elems (A) + tile slice (B)

  // ---- staging helpers: load phase issues all global loads back-to-back,
  //      store phase converts to f16 and writes LDS (one wait total) ----
  auto loadA = [&](int k0) {
#pragma unroll
    for (int it = 0; it < 2; ++it) {
      int idx = it * 128 + tid;       // 0..255
      int r   = idx >> 2;             // 0..63
      int c   = (idx & 3) << 3;       // 0,8,16,24
      const float* p = A + (long)(tile_m + r) * lda + (k0 + c);
      ra[2 * it + 0] = *(const v4f*)p;
      ra[2 * it + 1] = *(const v4f*)(p + 4);
    }
  };
  auto storeA = [&](int b) {
#pragma unroll
    for (int it = 0; it < 2; ++it) {
      int idx = it * 128 + tid;
      int r   = idx >> 2;
      int c   = (idx & 3) << 3;
      v4f lo = ra[2 * it + 0], hi = ra[2 * it + 1];
      _Float16 h8[8] = {(_Float16)lo.x, (_Float16)lo.y, (_Float16)lo.z, (_Float16)lo.w,
                        (_Float16)hi.x, (_Float16)hi.y, (_Float16)hi.z, (_Float16)hi.w};
      *(v4u*)(&As[b][r][c]) = *(const v4u*)h8;
    }
  };
  auto loadB = [&](int k0) {
    if (transB) {
#pragma unroll
      for (int it = 0; it < 2; ++it) {
        int idx = it * 128 + tid;
        int r   = idx >> 2;           // n
        int c   = (idx & 3) << 3;     // k
        const float* p = Bm + (long)(tile_n + r) * ldb + (k0 + c);
        rb[2 * it + 0] = *(const v4f*)p;
        rb[2 * it + 1] = *(const v4f*)(p + 4);
      }
    } else {
#pragma unroll
      for (int it = 0; it < 4; ++it) {
        int idx = it * 128 + tid;
        int r   = idx >> 4;           // k: 0..31
        int c   = (idx & 15) << 2;    // n: 0..60
        rb[it] = *(const v4f*)(Bm + (long)(k0 + r) * ldb + (tile_n + c));
      }
    }
  };
  auto storeB = [&](int b) {
    if (transB) {
#pragma unroll
      for (int it = 0; it < 2; ++it) {
        int idx = it * 128 + tid;
        int r   = idx >> 2;
        int c   = (idx & 3) << 3;
        v4f lo = rb[2 * it + 0], hi = rb[2 * it + 1];
        _Float16 h8[8] = {(_Float16)lo.x, (_Float16)lo.y, (_Float16)lo.z, (_Float16)lo.w,
                          (_Float16)hi.x, (_Float16)hi.y, (_Float16)hi.z, (_Float16)hi.w};
        *(v4u*)(&Bs[b][r][c]) = *(const v4u*)h8;
      }
    } else {
#pragma unroll
      for (int it = 0; it < 4; ++it) {
        int idx = it * 128 + tid;
        int r   = idx >> 4;
        int c   = (idx & 15) << 2;
        Bs[b][c + 0][r] = (_Float16)rb[it].x;
        Bs[b][c + 1][r] = (_Float16)rb[it].y;
        Bs[b][c + 2][r] = (_Float16)rb[it].z;
        Bs[b][c + 3][r] = (_Float16)rb[it].w;
      }
    }
  };

  // ---- pipeline prologue: stage tile 0 ----
  loadA(0); loadB(0);
  storeA(0); storeB(0);
  __syncthreads();

  int buf = 0;
  for (int k0 = 0; k0 < K; k0 += 32) {
    const bool has_next = (k0 + 32 < K);
    if (has_next) { loadA(k0 + 32); loadB(k0 + 32); }  // in flight under WMMA

    // ---- fragments per ISA 7.12.2 layouts ----
    Frag16 fa[2], fb[2];
#pragma unroll
    for (int i = 0; i < 2; ++i) {
      int r  = wrow + (i << 4) + l16;
      int kb = lhi << 3;                     // A: lanes<16 K0-7/16-23, >=16 K8-15/24-31
      fa[i].q[0] = *(const v4u*)(&As[buf][r][kb]);
      fa[i].q[1] = *(const v4u*)(&As[buf][r][kb + 16]);
      int n   = wcol + (i << 4) + l16;
      int kb2 = lhi << 3;                    // B: lanes<16 K0-7/16-23, >=16 K8-15/24-31
      fb[i].q[0] = *(const v4u*)(&Bs[buf][n][kb2]);
      fb[i].q[1] = *(const v4u*)(&Bs[buf][n][kb2 + 16]);
    }
#pragma unroll
    for (int i = 0; i < 2; ++i)
#pragma unroll
      for (int j = 0; j < 2; ++j)
        acc[i][j] = __builtin_amdgcn_wmma_f32_16x16x32_f16(
            false, fa[i].h, false, fb[j].h, (short)0, acc[i][j], false, false);

    if (has_next) { storeA(buf ^ 1); storeB(buf ^ 1); }
    __syncthreads();
    buf ^= 1;
  }

  // ---- fused epilogue: alpha, bias, residual, relu ----
#pragma unroll
  for (int i = 0; i < 2; ++i) {
#pragma unroll
    for (int j = 0; j < 2; ++j) {
      int mb = tile_m + wrow + (i << 4) + (lhi << 3);
      int nn = tile_n + wcol + (j << 4) + l16;
      float bv = bias ? bias[nn] : 0.0f;
#pragma unroll
      for (int r = 0; r < 8; ++r) {
        long off = (long)(mb + r) * ldc + nn;
        float v = alpha * acc[i][j][r] + bv;
        if (Rs) v += Rs[off];
        if (relu) v = fmaxf(v, 0.0f);
        C[off] = v;
      }
    }
  }
}

// ---------------------------------------------------------------------------
// Fused mask + softmax over the last axis of scores [B,H,Lq,Lk].
// mask_type 0: pad mask from ids (key pos).  1: pad OR causal (k > q).
// One wave per row (wave32 -> 16 elems/lane).
// ---------------------------------------------------------------------------
__global__ __launch_bounds__(256) void softmax_mask_kernel(
    float* __restrict__ sc, const int* __restrict__ ids, int mask_type)
{
  const int row  = blockIdx.x * 8 + (threadIdx.x >> 5);  // [0, B*H*L)
  const int lane = threadIdx.x & 31;
  const int b = row >> 12;        // row / (H*L)
  const int q = row & (L_SZ - 1); // row % L
  float* p = sc + (long)row * L_SZ;
  const int* idrow = ids + b * L_SZ;

  float v[16];
  float mx = -3.4e38f;
#pragma unroll
  for (int i = 0; i < 16; ++i) {
    int col = (i << 5) + lane;
    float s = p[col];
    bool masked = (idrow[col] == 0) || (mask_type == 1 && col > q);
    s += masked ? -1e9f : 0.0f;
    v[i] = s;
    mx = fmaxf(mx, s);
  }
#pragma unroll
  for (int o = 16; o > 0; o >>= 1) mx = fmaxf(mx, __shfl_xor(mx, o, 32));
  float sum = 0.0f;
#pragma unroll
  for (int i = 0; i < 16; ++i) { v[i] = expf(v[i] - mx); sum += v[i]; }
#pragma unroll
  for (int o = 16; o > 0; o >>= 1) sum += __shfl_xor(sum, o, 32);
  const float inv = 1.0f / sum;
#pragma unroll
  for (int i = 0; i < 16; ++i) p[(i << 5) + lane] = v[i] * inv;
}

// ---------------------------------------------------------------------------
// LayerNorm over rows of 512 (eps = 1e-3), one wave per row, two-pass.
// ---------------------------------------------------------------------------
__global__ __launch_bounds__(256) void layernorm_kernel(
    const float* __restrict__ in, const float* __restrict__ g,
    const float* __restrict__ be, float* __restrict__ out)
{
  const int row  = blockIdx.x * 8 + (threadIdx.x >> 5);
  const int lane = threadIdx.x & 31;
  const float* p = in + (long)row * D_SZ;

  float v[16]; float s = 0.0f;
#pragma unroll
  for (int i = 0; i < 16; ++i) { v[i] = p[(i << 5) + lane]; s += v[i]; }
#pragma unroll
  for (int o = 16; o > 0; o >>= 1) s += __shfl_xor(s, o, 32);
  const float mu = s * (1.0f / D_SZ);
  float s2 = 0.0f;
#pragma unroll
  for (int i = 0; i < 16; ++i) { float d = v[i] - mu; s2 += d * d; }
#pragma unroll
  for (int o = 16; o > 0; o >>= 1) s2 += __shfl_xor(s2, o, 32);
  const float inv = rsqrtf(s2 * (1.0f / D_SZ) + 1e-3f);
#pragma unroll
  for (int i = 0; i < 16; ++i) {
    int c = (i << 5) + lane;
    out[(long)row * D_SZ + c] = (v[i] - mu) * inv * g[c] + be[c];
  }
}

// ---------------------------------------------------------------------------
// Embedding gather + positional encoding: out[b,l,:] = table[ids[b,l]] + pe[l]
// ---------------------------------------------------------------------------
__global__ __launch_bounds__(128) void embed_kernel(
    const int* __restrict__ ids, const float* __restrict__ tab,
    const float* __restrict__ pe, float* __restrict__ out)
{
  const int row = blockIdx.x;           // [0, B*L)
  const int l   = row & (L_SZ - 1);
  const int id  = ids[row];
  const int c   = threadIdx.x << 2;
  v4f t  = *(const v4f*)(tab + (long)id * D_SZ + c);
  v4f pp = *(const v4f*)(pe + (long)l * D_SZ + c);
  *(v4f*)(out + (long)row * D_SZ + c) = t + pp;
}

// ---------------------------------------------------------------------------
// Host driver
// ---------------------------------------------------------------------------
struct MHp { const float *wq, *bq, *wk, *bk, *wv, *bv, *wo, *bo; };
struct LNp { const float *g, *b; };
struct FFp { const float *w1, *b1, *w2, *b2; };

extern "C" void kernel_launch(void* const* d_in, const int* in_sizes, int n_in,
                              void* d_out, int out_size, void* d_ws, size_t ws_size,
                              hipStream_t stream)
{
  (void)in_sizes; (void)n_in; (void)out_size; (void)ws_size;

  int ci = 0;
  auto nf = [&]() { return (const float*)d_in[ci++]; };

  const int*   x     = (const int*)d_in[ci++];
  const int*   y     = (const int*)d_in[ci++];
  const float* table = nf();
  const float* pe    = nf();

  MHp enc_mh[4]; LNp enc_ln1[4], enc_ln2[4]; FFp enc_ff[4];
  for (int l = 0; l < 4; ++l) {
    enc_mh[l]  = { nf(), nf(), nf(), nf(), nf(), nf(), nf(), nf() };
    enc_ln1[l] = { nf(), nf() };
    enc_ln2[l] = { nf(), nf() };
    enc_ff[l]  = { nf(), nf(), nf(), nf() };
  }
  MHp dec_mh1[4], dec_mh2[4]; LNp dec_ln1[4], dec_ln2[4], dec_ln3[4]; FFp dec_ff[4];
  for (int l = 0; l < 4; ++l) {
    dec_mh1[l] = { nf(), nf(), nf(), nf(), nf(), nf(), nf(), nf() };
    dec_mh2[l] = { nf(), nf(), nf(), nf(), nf(), nf(), nf(), nf() };
    dec_ln1[l] = { nf(), nf() };
    dec_ln2[l] = { nf(), nf() };
    dec_ln3[l] = { nf(), nf() };
    dec_ff[l]  = { nf(), nf(), nf(), nf() };
  }
  const float* Wout = nf();
  const float* bout = nf();

  // ---- workspace layout (f32), ~84 MB total ----
  float* ws = (float*)d_ws;
  const long U = 1L << 20;          // 1 Mi floats per activation buffer
  float* xz  = ws;
  float* yz  = ws + 1 * U;
  float* qb  = ws + 2 * U;
  float* kb  = ws + 3 * U;
  float* vb  = ws + 4 * U;
  float* ctx = ws + 5 * U;
  float* t0  = ws + 6 * U;
  float* t1  = ws + 7 * U;
  float* t2  = ws + 8 * U;
  float* mid = ws + 9 * U;          // 4 Mi floats [2048, 2048]
  float* sc  = ws + 13 * U;         // 8 Mi floats [B,H,L,L]

  auto gemm = [&](const float* A, const float* Bm, const float* bias,
                  const float* Rs, float* C, int Mm, int Nn, int Kk,
                  int lda, int ldb, int ldc, int nb, int bdiv,
                  long sA1, long sA2, long sB1, long sB2, long sC1, long sC2,
                  int transB, float alpha, int relu) {
    dim3 grid(Nn / 64, Mm / 64, nb), blk(128);
    gemm_wmma_f16<<<grid, blk, 0, stream>>>(A, Bm, bias, Rs, C, Kk, lda, ldb,
        ldc, sA1, sA2, sB1, sB2, sC1, sC2, bdiv, transB, alpha, relu);
  };
  auto lnorm = [&](const float* in, const LNp& p, float* out) {
    layernorm_kernel<<<dim3(M_SZ / 8), dim3(256), 0, stream>>>(in, p.g, p.b, out);
  };

  const float SCALE = 1.0f / (8.0f + 1e-8f);   // 1/(sqrt(64)+1e-8)
  const long  LD    = (long)L_SZ * L_SZ;       // 512*512

  auto attention = [&](const float* qin, const float* kvin, const MHp& m,
                       const int* mids, int mtype, const float* resid,
                       float* outb) {
    gemm(qin,  m.wq, m.bq, nullptr, qb, M_SZ, D_SZ, D_SZ, D_SZ, D_SZ, D_SZ,
         1, 1, 0, 0, 0, 0, 0, 0, 0, 1.0f, 0);
    gemm(kvin, m.wk, m.bk, nullptr, kb, M_SZ, D_SZ, D_SZ, D_SZ, D_SZ, D_SZ,
         1, 1, 0, 0, 0, 0, 0, 0, 0, 1.0f, 0);
    gemm(kvin, m.wv, m.bv, nullptr, vb, M_SZ, D_SZ, D_SZ, D_SZ, D_SZ, D_SZ,
         1, 1, 0, 0, 0, 0, 0, 0, 0, 1.0f, 0);
    // scores[b,h] = scale * Q_h @ K_h^T   (batched over 32 (b,h), transB)
    gemm(qb, kb, nullptr, nullptr, sc, L_SZ, L_SZ, HD_SZ, D_SZ, D_SZ, L_SZ,
         B_SZ * H_SZ, H_SZ, LD, HD_SZ, LD, HD_SZ, 8 * LD, LD, 1, SCALE, 0);
    softmax_mask_kernel<<<dim3(B_SZ * H_SZ * L_SZ / 8), dim3(256), 0, stream>>>(
        sc, mids, mtype);
    // ctx[b,h] = attn @ V_h
    gemm(sc, vb, nullptr, nullptr, ctx, L_SZ, HD_SZ, L_SZ, L_SZ, D_SZ, D_SZ,
         B_SZ * H_SZ, H_SZ, 8 * LD, LD, LD, HD_SZ, LD, HD_SZ, 0, 1.0f, 0);
    // out = ctx @ Wo + bo + residual
    gemm(ctx, m.wo, m.bo, resid, outb, M_SZ, D_SZ, D_SZ, D_SZ, D_SZ, D_SZ,
         1, 1, 0, 0, 0, 0, 0, 0, 0, 1.0f, 0);
  };

  // ---- embeddings + positional encoding ----
  embed_kernel<<<dim3(M_SZ), dim3(128), 0, stream>>>(x, table, pe, xz);
  embed_kernel<<<dim3(M_SZ), dim3(128), 0, stream>>>(y, table, pe, yz);

  // ---- encoder ----
  for (int l = 0; l < 4; ++l) {
    attention(xz, xz, enc_mh[l], x, 0, xz, t0);          // t0 = MHA + xz
    lnorm(t0, enc_ln1[l], t1);                           // o1
    gemm(t1, enc_ff[l].w1, enc_ff[l].b1, nullptr, mid, M_SZ, FF_SZ, D_SZ,
         D_SZ, FF_SZ, FF_SZ, 1, 1, 0, 0, 0, 0, 0, 0, 0, 1.0f, 1);
    gemm(mid, enc_ff[l].w2, enc_ff[l].b2, t1, t2, M_SZ, D_SZ, FF_SZ,
         FF_SZ, D_SZ, D_SZ, 1, 1, 0, 0, 0, 0, 0, 0, 0, 1.0f, 0);
    lnorm(t2, enc_ln2[l], xz);
  }

  // ---- decoder ----
  for (int l = 0; l < 4; ++l) {
    attention(yz, yz, dec_mh1[l], y, 1, yz, t0);         // causal self-attn
    lnorm(t0, dec_ln1[l], t1);                           // o1
    attention(t1, xz, dec_mh2[l], x, 0, t1, t0);         // cross-attn + o1
    lnorm(t0, dec_ln2[l], t2);                           // o2
    gemm(t2, dec_ff[l].w1, dec_ff[l].b1, nullptr, mid, M_SZ, FF_SZ, D_SZ,
         D_SZ, FF_SZ, FF_SZ, 1, 1, 0, 0, 0, 0, 0, 0, 0, 1.0f, 1);
    gemm(mid, dec_ff[l].w2, dec_ff[l].b2, t2, t0, M_SZ, D_SZ, FF_SZ,
         FF_SZ, D_SZ, D_SZ, 1, 1, 0, 0, 0, 0, 0, 0, 0, 1.0f, 0);
    lnorm(t0, dec_ln3[l], yz);
  }

  // ---- final projection to vocab logits (dominant GEMM: 2048x32000x512) ----
  gemm(yz, Wout, bout, nullptr, (float*)d_out, M_SZ, V_SZ, D_SZ,
       D_SZ, V_SZ, V_SZ, 1, 1, 0, 0, 0, 0, 0, 0, 0, 1.0f, 0);
}
